// FiLMGnn_56848187129857
// MI455X (gfx1250) — matrix operand order repaired and checked by reading, not compiled, not run on hardware
//
#include <hip/hip_runtime.h>
#include <hip/hip_fp16.h>

#define N_NODES 25000
#define M_PAD   25088            // N_NODES padded to multiple of 128
#define N_EDGES 400000
#define EF_EDGES (N_EDGES + N_NODES)
#define LAYERS 6

typedef __attribute__((ext_vector_type(16))) _Float16 v16h;
typedef __attribute__((ext_vector_type(8)))  float    v8f;
typedef __attribute__((ext_vector_type(4)))  _Float16 h4;

// ---------------- CDNA5 async global->LDS staging ----------------
__device__ __forceinline__ void async_copy_b128(void* lds_ptr, const void* gptr) {
  unsigned lds = (unsigned)(unsigned long long)lds_ptr;   // low 32 bits = LDS offset
  asm volatile("global_load_async_to_lds_b128 %0, %1, off"
               :: "v"(lds), "v"(gptr) : "memory");
}
__device__ __forceinline__ void async_wait0() {
  asm volatile("s_wait_asynccnt 0x0" ::: "memory");
}

// ---------------- WMMA helpers (CDNA5 16x16x32 f16 -> f32) ----------------
__device__ __forceinline__ v8f wmma16x16x32(v16h a, v16h b, v8f c) {
  return __builtin_amdgcn_wmma_f32_16x16x32_f16(
      false, a, false, b, (short)0, c, false, false);
}

// A fragment: 16x32 f16 tile, row-major in LDS, row stride lda halves.
__device__ __forceinline__ v16h load_a_frag(const _Float16* As, int lda, int k0, int lane) {
  const int m  = lane & 15;
  const int kb = k0 + ((lane & 16) ? 16 : 0);
  const _Float16* p = As + m * lda + kb;
  v16h a;
#pragma unroll
  for (int i = 0; i < 8; ++i) { a[i] = p[i]; a[i + 8] = p[i + 8]; }
  return a;
}

// B fragment: 32x16 f16, stored N-major in LDS (Bt[n][k], row stride ldb halves).
__device__ __forceinline__ v16h load_b_frag(const _Float16* Bt, int ldb, int ncol0, int k0, int lane) {
  const int n  = lane & 15;
  const int kb = k0 + ((lane & 16) ? 16 : 0);
  const _Float16* p = Bt + (ncol0 + n) * ldb + kb;
  v16h b;
#pragma unroll
  for (int i = 0; i < 16; ++i) b[i] = p[i];
  return b;
}

// ordered-float trick for atomicMax on floats
__device__ __forceinline__ unsigned f2ord(float f) {
  unsigned u = __float_as_uint(f);
  return (u & 0x80000000u) ? ~u : (u | 0x80000000u);
}
__device__ __forceinline__ float ord2f(unsigned u) {
  return (u & 0x80000000u) ? __uint_as_float(u & 0x7fffffffu) : __uint_as_float(~u);
}

// ---------------- f32 -> f16 conversions (done once; enables pure-copy async staging) ----------------
__global__ void cvt_f16_kernel(const float* __restrict__ in, _Float16* __restrict__ out, int n) {
  const int idx = blockIdx.x * blockDim.x + threadIdx.x;
  if (idx < n) out[idx] = (_Float16)in[idx];
}

// out[n*K + k] = W[k*N + n]  (N-major transpose, f16)
__global__ void transpose_f16_kernel(const float* __restrict__ W, _Float16* __restrict__ out,
                                     int K, int N) {
  const int idx = blockIdx.x * blockDim.x + threadIdx.x;
  if (idx >= K * N) return;
  const int n = idx / K, k = idx - n * K;
  out[idx] = (_Float16)W[(long long)k * N + n];
}

// ---------------- GEMM: C[M_PAD,N] = Ah[M_PAD,128] @ BtT^T, all-async staging ----------------
// Ah row-major f16; BtT is B pre-transposed N-major f16 [N x 128].
// grid = (M_PAD/128, N/128), block = 256 (8 waves; wave w owns rows w*16..w*16+15).
template <bool BIAS, bool HALFOUT>
__global__ __launch_bounds__(256)
void gemm_async_kernel(const _Float16* __restrict__ Ah, const _Float16* __restrict__ BtT,
                       const float* __restrict__ bias, float* __restrict__ C,
                       _Float16* __restrict__ Ch, int N) {
  __shared__ __align__(16) _Float16 As[128 * 128];   // 32 KB
  __shared__ __align__(16) _Float16 Bt[128 * 128];   // 32 KB
  const int tid = threadIdx.x;
  const long long mblk = (long long)blockIdx.x * 128;
  const long long nblk = (long long)blockIdx.y * 128;

  // A tile: rows mblk..mblk+127 are one contiguous 32KB region -> linear async copy.
  const _Float16* ga = Ah + mblk * 128;
  for (int i = tid; i < 2048; i += 256) async_copy_b128(As + i * 8, ga + i * 8);
  // B tile: rows nblk..nblk+127 of BtT, also contiguous.
  const _Float16* gb = BtT + nblk * 128;
  for (int i = tid; i < 2048; i += 256) async_copy_b128(Bt + i * 8, gb + i * 8);
  async_wait0();
  __syncthreads();

  const int wave = tid >> 5, lane = tid & 31;
  const _Float16* Aw = As + wave * 16 * 128;
#pragma unroll
  for (int nt = 0; nt < 8; ++nt) {
    v8f acc = {};
#pragma unroll
    for (int k0 = 0; k0 < 128; k0 += 32) {
      v16h a = load_a_frag(Aw, 128, k0, lane);
      v16h b = load_b_frag(Bt, 128, nt * 16, k0, lane);
      acc = wmma16x16x32(a, b, acc);
    }
    const int ncol = (int)nblk + nt * 16 + (lane & 15);
    const int mo   = (lane & 16) ? 8 : 0;
    const float bv = BIAS ? bias[ncol] : 0.f;
#pragma unroll
    for (int r = 0; r < 8; ++r) {
      const long long gm = mblk + wave * 16 + mo + r;
      const float v = acc[r] + bv;
      C[gm * N + ncol] = v;
      if (HALFOUT) Ch[gm * N + ncol] = (_Float16)v;
    }
  }
}

// ---------------- edge_attr column sums (self-loop fill_value='mean') ----------------
__global__ void ea_sum_kernel(const float* __restrict__ ea, float* __restrict__ sum) {
  const int tid = blockIdx.x * blockDim.x + threadIdx.x;
  const int j = tid & 31;
  const int stride = (gridDim.x * blockDim.x) >> 5;
  float s = 0.f;
  for (int r = tid >> 5; r < N_EDGES; r += stride) s += ea[(long long)r * 32 + j];
  atomicAdd(&sum[j], s);
}

// ---------------- LayerNorm + FiLM (one wave per node) -> f16 output ----------------
__global__ __launch_bounds__(256)
void lnfilm_kernel(const float* __restrict__ h, const float* __restrict__ ls,
                   const float* __restrict__ lb, const float* __restrict__ gamma,
                   const float* __restrict__ beta, _Float16* __restrict__ hmh) {
  const int wave = threadIdx.x >> 5, lane = threadIdx.x & 31;
  const int n = blockIdx.x * 8 + wave;
  if (n >= N_NODES) return;
  float v[4];
  const float4 v4 = *(const float4*)(h + (long long)n * 128 + lane * 4);
  v[0] = v4.x; v[1] = v4.y; v[2] = v4.z; v[3] = v4.w;
  float s = v[0] + v[1] + v[2] + v[3];
  float s2 = v[0]*v[0] + v[1]*v[1] + v[2]*v[2] + v[3]*v[3];
#pragma unroll
  for (int m = 16; m; m >>= 1) { s += __shfl_xor(s, m, 32); s2 += __shfl_xor(s2, m, 32); }
  const float mu = s * (1.f / 128.f);
  const float var = s2 * (1.f / 128.f) - mu * mu;
  const float rs = rsqrtf(var + 1e-5f);
  h4 o;
#pragma unroll
  for (int j = 0; j < 4; ++j) {
    const int d = lane * 4 + j;
    o[j] = (_Float16)(((v[j] - mu) * rs * ls[d] + lb[d]) * gamma[d] + beta[d]);
  }
  *(h4*)(hmh + (long long)n * 128 + lane * 4) = o;
}

// ---------------- precompute u_src = W.a_src, u_dst = W.a_dst, v_e = We.a_edge ----------------
__global__ void attvec_kernel(const float* __restrict__ W, const float* __restrict__ asv,
                              const float* __restrict__ adv, const float* __restrict__ We,
                              const float* __restrict__ aev,
                              float* __restrict__ u_src, float* __restrict__ u_dst,
                              float* __restrict__ v_e) {
  const int idx = blockIdx.x * blockDim.x + threadIdx.x;
  if (idx < 512) {
    const int d = idx >> 2, hh = idx & 3;
    float s = 0.f, t = 0.f;
    for (int k = 0; k < 128; ++k) {
      const float w = W[(long long)d * 512 + hh * 128 + k];
      s += w * asv[hh * 128 + k];
      t += w * adv[hh * 128 + k];
    }
    u_src[idx] = s; u_dst[idx] = t;
  } else if (idx < 512 + 128) {
    const int j = idx - 512, ef = j >> 2, hh = j & 3;
    float s = 0.f;
    for (int k = 0; k < 128; ++k) s += We[(long long)ef * 512 + hh * 128 + k] * aev[hh * 128 + k];
    v_e[j] = s;
  }
}

// ---------------- per-node attention scores from f16 hm ----------------
__global__ void node_score_kernel(const _Float16* __restrict__ hmh, const float* __restrict__ u_src,
                                  const float* __restrict__ u_dst,
                                  float* __restrict__ al_s, float* __restrict__ al_d) {
  const int idx = blockIdx.x * blockDim.x + threadIdx.x;
  if (idx >= N_NODES * 4) return;
  const int n = idx >> 2, hh = idx & 3;
  const _Float16* row = hmh + (long long)n * 128;
  float s = 0.f, t = 0.f;
  for (int d = 0; d < 128; ++d) {
    const float v = (float)row[d];
    s += v * u_src[d * 4 + hh];
    t += v * u_dst[d * 4 + hh];
  }
  al_s[idx] = s; al_d[idx] = t;
}

// ---------------- raw alpha + segment max ----------------
__global__ void alpha_raw_kernel(const float* __restrict__ ea, const float* __restrict__ easum,
                                 const float* __restrict__ v_e,
                                 const int* __restrict__ srcI, const int* __restrict__ dstI,
                                 const float* __restrict__ al_s, const float* __restrict__ al_d,
                                 float* __restrict__ alpha, unsigned* __restrict__ amax) {
  const int idx = blockIdx.x * blockDim.x + threadIdx.x;
  if (idx >= EF_EDGES * 4) return;
  const int e = idx >> 2, hh = idx & 3;
  int s, d;
  float ae = 0.f;
  if (e < N_EDGES) {
    s = srcI[e]; d = dstI[e];
    const float* row = ea + (long long)e * 32;
#pragma unroll
    for (int j = 0; j < 32; ++j) ae += row[j] * v_e[j * 4 + hh];
  } else {
    s = d = e - N_EDGES;
    const float inv = 1.f / (float)N_EDGES;
#pragma unroll
    for (int j = 0; j < 32; ++j) ae += easum[j] * inv * v_e[j * 4 + hh];
  }
  float a = al_s[s * 4 + hh] + al_d[d * 4 + hh] + ae;
  a = (a > 0.f) ? a : 0.2f * a;           // leaky_relu(0.2)
  alpha[idx] = a;
  atomicMax(&amax[d * 4 + hh], f2ord(a));
}

// ---------------- exp(alpha - max) + segment sum ----------------
__global__ void alpha_exp_kernel(const int* __restrict__ dstI, float* __restrict__ alpha,
                                 const unsigned* __restrict__ amax, float* __restrict__ denom) {
  const int idx = blockIdx.x * blockDim.x + threadIdx.x;
  if (idx >= EF_EDGES * 4) return;
  const int e = idx >> 2, hh = idx & 3;
  const int d = (e < N_EDGES) ? dstI[e] : (e - N_EDGES);
  const float a = __expf(alpha[idx] - ord2f(amax[d * 4 + hh]));
  alpha[idx] = a;
  atomicAdd(&denom[d * 4 + hh], a);
}

// ---------------- message scatter: out[dst] += x[src] * alpha ----------------
__global__ __launch_bounds__(128)
void message_kernel(const int* __restrict__ srcI, const int* __restrict__ dstI,
                    const float* __restrict__ x, const float* __restrict__ alpha,
                    const float* __restrict__ denom, float* __restrict__ oacc) {
  const int e = blockIdx.x, t = threadIdx.x;
  int s, d;
  if (e < N_EDGES) { s = srcI[e]; d = dstI[e]; } else { s = d = e - N_EDGES; }
  const float* xs = x + (long long)s * 512;
  float* od = oacc + (long long)d * 512;
  __builtin_prefetch(xs + t, 0, 0);   // global_prefetch_b8 (x buffer is L2-resident)
#pragma unroll
  for (int hh = 0; hh < 4; ++hh) {
    const float w = alpha[e * 4 + hh] / (denom[d * 4 + hh] + 1e-16f);
    const int c = hh * 128 + t;
    atomicAdd(&od[c], xs[c] * w);
  }
}

// ---------------- head-mean + bias + relu; writes f32 h and f16 h ----------------
__global__ void finish_kernel(const float* __restrict__ oacc, const float* __restrict__ bias,
                              float* __restrict__ h, _Float16* __restrict__ hhf) {
  const int idx = blockIdx.x * blockDim.x + threadIdx.x;
  if (idx >= N_NODES * 128) return;
  const int n = idx >> 7, c = idx & 127;
  const float* o = oacc + (long long)n * 512;
  const float s = fmaxf(0.25f * (o[c] + o[128 + c] + o[256 + c] + o[384 + c]) + bias[c], 0.f);
  h[idx] = s;
  hhf[idx] = (_Float16)s;
}

// ---------------- edge head: relu([h_s,h_d,ea] @ W1 + b1) @ W2 + b2 ----------------
// block = 128 (4 waves); wave owns 16 edges. K=288, N chunked 4x32. W1tT = W1^T f16 [128x288].
__global__ __launch_bounds__(128)
void head_kernel(const _Float16* __restrict__ hhf, const float* __restrict__ ea,
                 const int* __restrict__ srcI, const int* __restrict__ dstI,
                 const _Float16* __restrict__ W1tT, const float* __restrict__ b1,
                 const float* __restrict__ W2, const float* __restrict__ b2,
                 float* __restrict__ out) {
  __shared__ __align__(16) _Float16 Aty[4 * 16 * 288];   // 36 KB
  __shared__ __align__(16) _Float16 W1t[32 * 288];       // 18 KB
  __shared__ float lacc[64];
  const int tid = threadIdx.x, wave = tid >> 5, lane = tid & 31;
  const int e0 = blockIdx.x * 64 + wave * 16;
  _Float16* Aw = Aty + wave * 16 * 288;

  // async row-gather: cols 0-127 = h16[src], 128-255 = h16[dst] (16B chunks)
  for (int i = lane; i < 512; i += 32) {
    const int r = i >> 5, c = i & 31;
    const int e = e0 + r;
    _Float16* lp = Aw + r * 288 + c * 8;
    if (e < N_EDGES) {
      const long long node = (c < 16) ? (long long)srcI[e] : (long long)dstI[e];
      async_copy_b128(lp, hhf + node * 128 + (c & 15) * 8);
    } else {
#pragma unroll
      for (int j = 0; j < 8; ++j) lp[j] = (_Float16)0.f;
    }
  }
  // cols 256-287 = edge_attr (f32 -> f16)
  for (int i = lane; i < 512; i += 32) {
    const int r = i >> 5, j = i & 31;
    const int e = e0 + r;
    Aw[r * 288 + 256 + j] = (e < N_EDGES) ? (_Float16)ea[(long long)e * 32 + j] : (_Float16)0.f;
  }
  if (tid < 64) lacc[tid] = 0.f;
  async_wait0();
  __syncthreads();

  for (int nc = 0; nc < 4; ++nc) {
    // stage W1^T rows nc*32..+31 (contiguous) via async copy
    const _Float16* gw = W1tT + (long long)nc * 32 * 288;
    for (int i = tid; i < 1152; i += 128) async_copy_b128(W1t + i * 8, gw + i * 8);
    async_wait0();
    __syncthreads();
#pragma unroll
    for (int sub = 0; sub < 2; ++sub) {
      v8f acc = {};
#pragma unroll
      for (int k0 = 0; k0 < 288; k0 += 32) {
        v16h a = load_a_frag(Aw, 288, k0, lane);
        v16h b = load_b_frag(W1t, 288, sub * 16, k0, lane);
        acc = wmma16x16x32(a, b, acc);
      }
      const int n = nc * 32 + sub * 16 + (lane & 15);
      const float bv = b1[n], wv = W2[n];
      const int mo = (lane & 16) ? 8 : 0;
#pragma unroll
      for (int r = 0; r < 8; ++r) {
        const float v = fmaxf(acc[r] + bv, 0.f);
        atomicAdd(&lacc[wave * 16 + mo + r], v * wv);   // fused 128->1 projection
      }
    }
    __syncthreads();
  }
  if (lane < 16) {
    const int e = e0 + lane;
    if (e < N_EDGES) out[e] = lacc[wave * 16 + lane] + b2[0];
  }
}

// ---------------- host launcher ----------------
extern "C" void kernel_launch(void* const* d_in, const int* in_sizes, int n_in,
                              void* d_out, int out_size, void* d_ws, size_t ws_size,
                              hipStream_t stream) {
  (void)in_sizes; (void)n_in; (void)out_size; (void)ws_size;
  const float* x     = (const float*)d_in[0];
  const int*   ei    = (const int*)d_in[1];
  const float* ea    = (const float*)d_in[2];
  const float* gamma = (const float*)d_in[3];
  const float* beta  = (const float*)d_in[4];
  const float* embW  = (const float*)d_in[5];
  const float* embB  = (const float*)d_in[6];
  const float* lnS   = (const float*)d_in[7];
  const float* lnB   = (const float*)d_in[8];
  const float* gatW  = (const float*)d_in[9];
  const float* attS  = (const float*)d_in[10];
  const float* attD  = (const float*)d_in[11];
  const float* gatWe = (const float*)d_in[12];
  const float* attE  = (const float*)d_in[13];
  const float* gatB  = (const float*)d_in[14];
  const float* hW1   = (const float*)d_in[15];
  const float* hb1   = (const float*)d_in[16];
  const float* hW2   = (const float*)d_in[17];
  const float* hb2   = (const float*)d_in[18];
  float* out = (float*)d_out;
  const int* srcI = ei;
  const int* dstI = ei + N_EDGES;

  char* wp = (char*)d_ws;
  auto alloc = [&](size_t bytes) -> void* {
    void* p = (void*)wp;
    wp += (bytes + 255) & ~(size_t)255;
    return p;
  };
  float*     h     = (float*)alloc((size_t)M_PAD * 128 * 4);
  _Float16*  h16   = (_Float16*)alloc((size_t)M_PAD * 128 * 2);
  _Float16*  hm16  = (_Float16*)alloc((size_t)M_PAD * 128 * 2);
  _Float16*  x16   = (_Float16*)alloc((size_t)M_PAD * 128 * 2);
  float*     xb    = (float*)alloc((size_t)M_PAD * 512 * 4);
  float*     oacc  = (float*)alloc((size_t)M_PAD * 512 * 4);
  float*     al_s  = (float*)alloc((size_t)N_NODES * 4 * 4);
  float*     al_d  = (float*)alloc((size_t)N_NODES * 4 * 4);
  float*     alpha = (float*)alloc((size_t)EF_EDGES * 4 * 4);
  unsigned*  amax  = (unsigned*)alloc((size_t)N_NODES * 4 * 4);
  float*     denom = (float*)alloc((size_t)N_NODES * 4 * 4);
  float*     easum = (float*)alloc(32 * 4);
  float*     u_src = (float*)alloc(512 * 4);
  float*     u_dst = (float*)alloc(512 * 4);
  float*     v_e   = (float*)alloc(128 * 4);
  _Float16*  embWT = (_Float16*)alloc((size_t)128 * 128 * 2);
  _Float16*  gatWT = (_Float16*)alloc((size_t)LAYERS * 512 * 128 * 2);
  _Float16*  W1tT  = (_Float16*)alloc((size_t)128 * 288 * 2);

  // --- one-time f16 conversions / weight transposes (pure-copy async staging later) ---
  cvt_f16_kernel<<<(N_NODES * 128 + 255) / 256, 256, 0, stream>>>(x, x16, N_NODES * 128);
  transpose_f16_kernel<<<(128 * 128 + 255) / 256, 256, 0, stream>>>(embW, embWT, 128, 128);
  for (int l = 0; l < LAYERS; ++l)
    transpose_f16_kernel<<<(128 * 512 + 255) / 256, 256, 0, stream>>>(
        gatW + (size_t)l * 128 * 512, gatWT + (size_t)l * 512 * 128, 128, 512);
  transpose_f16_kernel<<<(288 * 128 + 255) / 256, 256, 0, stream>>>(hW1, W1tT, 288, 128);

  hipMemsetAsync(easum, 0, 32 * 4, stream);
  ea_sum_kernel<<<512, 256, 0, stream>>>(ea, easum);

  // h = x @ embed_W + embed_b (also emits f16 copy for the head gather)
  gemm_async_kernel<true, true><<<dim3(M_PAD / 128, 1), 256, 0, stream>>>(
      x16, embWT, embB, h, h16, 128);

  for (int l = 0; l < LAYERS; ++l) {
    lnfilm_kernel<<<(N_NODES + 7) / 8, 256, 0, stream>>>(
        h, lnS + l * 128, lnB + l * 128, gamma + l * 128, beta + l * 128, hm16);
    attvec_kernel<<<3, 256, 0, stream>>>(
        gatW + (size_t)l * 128 * 512, attS + (size_t)l * 512, attD + (size_t)l * 512,
        gatWe + (size_t)l * 32 * 512, attE + (size_t)l * 512, u_src, u_dst, v_e);
    gemm_async_kernel<false, false><<<dim3(M_PAD / 128, 4), 256, 0, stream>>>(
        hm16, gatWT + (size_t)l * 512 * 128, nullptr, xb, nullptr, 512);
    node_score_kernel<<<(N_NODES * 4 + 255) / 256, 256, 0, stream>>>(hm16, u_src, u_dst, al_s, al_d);
    hipMemsetAsync(amax, 0, (size_t)N_NODES * 4 * 4, stream);
    hipMemsetAsync(denom, 0, (size_t)N_NODES * 4 * 4, stream);
    hipMemsetAsync(oacc, 0, (size_t)M_PAD * 512 * 4, stream);
    alpha_raw_kernel<<<(EF_EDGES * 4 + 255) / 256, 256, 0, stream>>>(
        ea, easum, v_e, srcI, dstI, al_s, al_d, alpha, amax);
    alpha_exp_kernel<<<(EF_EDGES * 4 + 255) / 256, 256, 0, stream>>>(dstI, alpha, amax, denom);
    message_kernel<<<EF_EDGES, 128, 0, stream>>>(srcI, dstI, xb, alpha, denom, oacc);
    finish_kernel<<<(N_NODES * 128 + 255) / 256, 256, 0, stream>>>(oacc, gatB + l * 128, h, h16);
  }

  head_kernel<<<(N_EDGES + 63) / 64, 128, 0, stream>>>(
      h16, ea, srcI, dstI, W1tT, hb1, hW2, hb2, out);
}
